// GGNAdapter_28295244546287
// MI455X (gfx1250) — matrix operand order repaired and checked by reading, not compiled, not run on hardware
//
#include <hip/hip_runtime.h>
#include <hip/hip_bf16.h>
#include <stdint.h>

#define Hd 64
#define Sd 24
#define KIN (Sd * Hd) /* 1536 */

typedef __attribute__((ext_vector_type(8)))  float        v8f;
typedef __attribute__((ext_vector_type(8)))  unsigned int v8u;
typedef __attribute__((ext_vector_type(16))) __bf16       v16bf;

// bf16 weight arena offsets (elements)
#define OFF_ENC 0
#define OFF_TW1 98304
#define OFF_TW2 106496
#define OFF_MW1 114688
#define OFF_MW2 126976
#define OFF_UW1 133120
#define OFF_UW2 157696
#define WTOT    169984

#define CW 4  /* waves per block, encoder (each wave: 32-node tile) */
#define EW 8  /* waves per block, edge      */
#define UW 8  /* waves per block, update    */

static __device__ __forceinline__ unsigned short f2bf(float f) {
  return (unsigned short)((__float_as_uint(f) + 0x8000u) >> 16);
}
// pack two fp32 -> packed bf16 pair with round-to-nearest via v_perm_b32
static __device__ __forceinline__ unsigned bfpack(float lo, float hi) {
  unsigned a = __float_as_uint(lo) + 0x8000u;
  unsigned b = __float_as_uint(hi) + 0x8000u;
  return __builtin_amdgcn_perm(b, a, 0x07060302u);
}
static __device__ __forceinline__ v16bf asbf(v8u u) {
  union { v8u u; v16bf b; } c; c.u = u; return c.b;
}
static __device__ __forceinline__ float sigm(float x) { return 1.0f / (1.0f + __expf(-x)); }
static __device__ __forceinline__ float siluf(float x) { return x * sigm(x); }
static __device__ __forceinline__ void wsync() { __builtin_amdgcn_wave_barrier(); }

// A fragment (16x32 bf16). ISA layout: lane=(half,M); v0-3: K=8*half+0..7, v4-7: K=16+8*half+0..7
static __device__ __forceinline__ v16bf a_from_f32(const float* row, int kbase, int half) {
  const float4* p0 = (const float4*)(row + kbase + 8 * half);
  const float4* p1 = (const float4*)(row + kbase + 16 + 8 * half);
  float4 a0 = p0[0], a1 = p0[1], b0 = p1[0], b1 = p1[1];
  v8u u;
  u[0] = bfpack(a0.x, a0.y); u[1] = bfpack(a0.z, a0.w);
  u[2] = bfpack(a1.x, a1.y); u[3] = bfpack(a1.z, a1.w);
  u[4] = bfpack(b0.x, b0.y); u[5] = bfpack(b0.z, b0.w);
  u[6] = bfpack(b1.x, b1.y); u[7] = bfpack(b1.z, b1.w);
  return asbf(u);
}
static __device__ __forceinline__ v16bf a_from_bf16(const unsigned short* row, int kbase, int half) {
  const uint4* p0 = (const uint4*)(row + kbase + 8 * half);
  const uint4* p1 = (const uint4*)(row + kbase + 16 + 8 * half);
  uint4 a = p0[0], b = p1[0];
  v8u u;
  u[0] = a.x; u[1] = a.y; u[2] = a.z; u[3] = a.w;
  u[4] = b.x; u[5] = b.y; u[6] = b.z; u[7] = b.w;
  return asbf(u);
}
// B fragment (32x16): B[k][n] = W[nbase+n][kbase+k]; lane=(khalf,n); v0-3: K=8*khalf+0..7, v4-7: K=16+8*khalf+0..7
static __device__ __forceinline__ v16bf b_from_w(const unsigned short* w, int ldk,
                                                 int nbase, int kbase, int n, int khalf) {
  const unsigned short* p = w + (size_t)(nbase + n) * ldk + kbase + khalf * 8;
  uint4 a = *(const uint4*)p, b = *(const uint4*)(p + 16);
  v8u u;
  u[0] = a.x; u[1] = a.y; u[2] = a.z; u[3] = a.w;
  u[4] = b.x; u[5] = b.y; u[6] = b.z; u[7] = b.w;
  return asbf(u);
}
static __device__ __forceinline__ v8f wmma_bf16(v16bf a, v16bf b, v8f c) {
  return __builtin_amdgcn_wmma_f32_16x16x32_bf16(false, a, false, b, (short)0, c, false, false);
}
// Write C fragments (16 x NT*16) as bf16 row-major tile of width W into LDS
template <int NT, int W>
static __device__ __forceinline__ void stage(unsigned short* sh, const v8f* c, int half, int n) {
#pragma unroll
  for (int nt = 0; nt < NT; ++nt)
#pragma unroll
    for (int r = 0; r < 8; ++r)
      sh[(r + 8 * half) * W + nt * 16 + n] = f2bf(c[nt][r]);
}
// cooperative global(bf16) -> LDS copy, elems multiple of 8
static __device__ __forceinline__ void ldscpy(unsigned short* dst, const unsigned short* src,
                                              int elems, int tid, int nthreads) {
  const uint4* s = (const uint4*)src;
  uint4* d = (uint4*)dst;
  int n16 = elems >> 3;
  for (int i = tid; i < n16; i += nthreads) d[i] = s[i];
}

__global__ void k_prep(unsigned short* __restrict__ wbf,
                       const float* __restrict__ encW, const float* __restrict__ tw1,
                       const float* __restrict__ tw2,  const float* __restrict__ mw1,
                       const float* __restrict__ mw2,  const float* __restrict__ uw1,
                       const float* __restrict__ uw2) {
  int i = blockIdx.x * blockDim.x + threadIdx.x;
  if (i >= WTOT) return;
  float v;
  if      (i < OFF_TW1) v = encW[i - OFF_ENC];
  else if (i < OFF_TW2) v = tw1[i - OFF_TW1];
  else if (i < OFF_MW1) v = tw2[i - OFF_TW2];
  else if (i < OFF_MW2) v = mw1[i - OFF_MW1];
  else if (i < OFF_UW1) v = mw2[i - OFF_MW2];
  else if (i < OFF_UW2) v = uw1[i - OFF_UW1];
  else                  v = uw2[i - OFF_UW2];
  wbf[i] = f2bf(v);
}

// Encoder GEMM (K=1536) + two temporal MLP blocks.
// CW waves/block; each wave owns a 32-node tile (two 16-row A tiles share every B fragment).
__global__ __launch_bounds__(32 * CW) void k_enc(const float* __restrict__ x,
                                                 const unsigned short* __restrict__ wbf,
                                                 const float* __restrict__ encB,
                                                 const float* __restrict__ tb1,
                                                 const float* __restrict__ tb2,
                                                 float* __restrict__ h,
                                                 unsigned short* __restrict__ hb, int N) {
  __shared__ __align__(16) unsigned short shs[CW][2 * 16 * 64];
  const int tid = threadIdx.x, wave = tid >> 5, lane = tid & 31;
  const int half = lane >> 4, n = lane & 15;
  unsigned short* sh = shs[wave];
  const int node0 = (blockIdx.x * CW + wave) * 32;
  int mr0 = node0 + n;      if (mr0 >= N) mr0 = N - 1;
  int mr1 = node0 + 16 + n; if (mr1 >= N) mr1 = N - 1;
  const float* xr0 = x + (size_t)mr0 * KIN;
  const float* xr1 = x + (size_t)mr1 * KIN;
  const unsigned short* encw = wbf + OFF_ENC;

  v8f acc[2][4] = {};
  for (int k0 = 0; k0 < KIN; k0 += 32) {
    v16bf a0 = a_from_f32(xr0, k0, half);
    v16bf a1 = a_from_f32(xr1, k0, half);
#pragma unroll
    for (int nt = 0; nt < 4; ++nt) {
      v16bf b = b_from_w(encw, KIN, nt * 16, k0, n, half);
      acc[0][nt] = wmma_bf16(a0, b, acc[0][nt]);
      acc[1][nt] = wmma_bf16(a1, b, acc[1][nt]);
    }
  }
#pragma unroll
  for (int nt = 0; nt < 4; ++nt) {
    float bia = encB[nt * 16 + n];
#pragma unroll
    for (int m = 0; m < 2; ++m)
#pragma unroll
      for (int r = 0; r < 8; ++r) acc[m][nt][r] += bia;
  }

  for (int t = 0; t < 2; ++t) {
    const unsigned short* w1 = wbf + OFF_TW1 + t * 4096;
    const unsigned short* w2 = wbf + OFF_TW2 + t * 4096;
    // ---- layer 1: relu(h @ W1^T + b1)
    wsync();
    stage<4, 64>(sh, acc[0], half, n);
    stage<4, 64>(sh + 16 * 64, acc[1], half, n);
    wsync();
#pragma unroll
    for (int m = 0; m < 2; ++m)
#pragma unroll
      for (int nt = 0; nt < 4; ++nt) acc[m][nt] = (v8f){};
#pragma unroll
    for (int c = 0; c < 2; ++c) {
      v16bf a0 = a_from_bf16(sh + n * 64, c * 32, half);
      v16bf a1 = a_from_bf16(sh + 16 * 64 + n * 64, c * 32, half);
#pragma unroll
      for (int nt = 0; nt < 4; ++nt) {
        v16bf b = b_from_w(w1, 64, nt * 16, c * 32, n, half);
        acc[0][nt] = wmma_bf16(a0, b, acc[0][nt]);
        acc[1][nt] = wmma_bf16(a1, b, acc[1][nt]);
      }
    }
#pragma unroll
    for (int nt = 0; nt < 4; ++nt) {
      float bia = tb1[t * 64 + nt * 16 + n];
#pragma unroll
      for (int m = 0; m < 2; ++m)
#pragma unroll
        for (int r = 0; r < 8; ++r) acc[m][nt][r] = fmaxf(acc[m][nt][r] + bia, 0.0f);
    }
    // ---- layer 2: (.) @ W2^T + b2
    wsync();
    stage<4, 64>(sh, acc[0], half, n);
    stage<4, 64>(sh + 16 * 64, acc[1], half, n);
    wsync();
#pragma unroll
    for (int m = 0; m < 2; ++m)
#pragma unroll
      for (int nt = 0; nt < 4; ++nt) acc[m][nt] = (v8f){};
#pragma unroll
    for (int c = 0; c < 2; ++c) {
      v16bf a0 = a_from_bf16(sh + n * 64, c * 32, half);
      v16bf a1 = a_from_bf16(sh + 16 * 64 + n * 64, c * 32, half);
#pragma unroll
      for (int nt = 0; nt < 4; ++nt) {
        v16bf b = b_from_w(w2, 64, nt * 16, c * 32, n, half);
        acc[0][nt] = wmma_bf16(a0, b, acc[0][nt]);
        acc[1][nt] = wmma_bf16(a1, b, acc[1][nt]);
      }
    }
#pragma unroll
    for (int nt = 0; nt < 4; ++nt) {
      float bia = tb2[t * 64 + nt * 16 + n];
#pragma unroll
      for (int m = 0; m < 2; ++m)
#pragma unroll
        for (int r = 0; r < 8; ++r) acc[m][nt][r] += bia;
    }
  }

#pragma unroll
  for (int m = 0; m < 2; ++m)
#pragma unroll
    for (int nt = 0; nt < 4; ++nt)
#pragma unroll
      for (int r = 0; r < 8; ++r) {
        int row = node0 + m * 16 + r + 8 * half;
        if (row < N) {
          size_t o = (size_t)row * 64 + nt * 16 + n;
          float v = acc[m][nt][r];
          h[o] = v; hb[o] = f2bf(v);
        }
      }
}

__global__ void k_zero(float4* __restrict__ p, long long n4) {
  long long i = blockIdx.x * (long long)blockDim.x + threadIdx.x;
  long long stride = gridDim.x * (long long)blockDim.x;
  float4 z = {0.f, 0.f, 0.f, 0.f};
  for (; i < n4; i += stride) p[i] = z;
}

// Edge message MLP + gate + atomic scatter; EW waves/block, one 16-edge tile per wave.
// msg weights staged to LDS once per block.
__global__ __launch_bounds__(32 * EW) void k_edge(const unsigned short* __restrict__ wbf,
                                                  const unsigned short* __restrict__ hb,
                                                  const int* __restrict__ srcI,
                                                  const int* __restrict__ dstI,
                                                  const float* __restrict__ mb1,
                                                  const float* __restrict__ mb2,
                                                  const float* __restrict__ gw,
                                                  const float* __restrict__ gb,
                                                  float* __restrict__ agg, int E, int s) {
  __shared__ __align__(16) unsigned short swm1[32 * 128];
  __shared__ __align__(16) unsigned short swm2[64 * 32];
  __shared__ __align__(16) unsigned short sms[EW][16 * 32];
  const int tid = threadIdx.x, wave = tid >> 5, lane = tid & 31;
  const int half = lane >> 4, n = lane & 15;
  unsigned short* sm = sms[wave];

  ldscpy(swm1, wbf + OFF_MW1 + s * (32 * 128), 32 * 128, tid, 32 * EW);
  ldscpy(swm2, wbf + OFF_MW2 + s * (64 * 32), 64 * 32, tid, 32 * EW);
  __syncthreads();

  const int e0 = (blockIdx.x * EW + wave) * 16;
  int e = e0 + n; int ec = (e < E) ? e : (E - 1);
  const int d = dstI[ec], sj = srcI[ec];

  v8f m1[2] = {};
#pragma unroll
  for (int kk = 0; kk < 4; ++kk) {
    const unsigned short* hrow = hb + (size_t)((kk < 2) ? d : sj) * 64;
    v16bf a = a_from_bf16(hrow, (kk & 1) * 32, half);
#pragma unroll
    for (int nt = 0; nt < 2; ++nt)
      m1[nt] = wmma_bf16(a, b_from_w(swm1, 128, nt * 16, kk * 32, n, half), m1[nt]);
  }
#pragma unroll
  for (int nt = 0; nt < 2; ++nt) {
    float bia = mb1[s * 32 + nt * 16 + n];
#pragma unroll
    for (int r = 0; r < 8; ++r) m1[nt][r] = siluf(m1[nt][r] + bia);
  }
  wsync();
  stage<2, 32>(sm, m1, half, n);
  wsync();

  v8f m2[4] = {};
  {
    v16bf a = a_from_bf16(sm + n * 32, 0, half);
#pragma unroll
    for (int nt = 0; nt < 4; ++nt)
      m2[nt] = wmma_bf16(a, b_from_w(swm2, 32, nt * 16, 0, n, half), m2[nt]);
  }
#pragma unroll
  for (int nt = 0; nt < 4; ++nt) {
    float bia = mb2[s * 64 + nt * 16 + n];
#pragma unroll
    for (int r = 0; r < 8; ++r) m2[nt][r] = siluf(m2[nt][r] + bia);
  }

  float gwv[4];
#pragma unroll
  for (int nt = 0; nt < 4; ++nt) gwv[nt] = gw[s * 64 + nt * 16 + n];
  float part[8];
#pragma unroll
  for (int r = 0; r < 8; ++r)
    part[r] = m2[0][r] * gwv[0] + m2[1][r] * gwv[1] + m2[2][r] * gwv[2] + m2[3][r] * gwv[3];
#pragma unroll
  for (int m = 1; m <= 8; m <<= 1)
#pragma unroll
    for (int r = 0; r < 8; ++r) part[r] += __shfl_xor(part[r], m, 32);

  const float gbv = gb[s];
#pragma unroll
  for (int r = 0; r < 8; ++r) {
    int erow = e0 + r + 8 * half;
    float g = (erow < E) ? sigm(part[r] + gbv) : 0.0f;
    int drow = __shfl(d, r + 8 * half, 32);
#pragma unroll
    for (int nt = 0; nt < 4; ++nt)
      atomicAdd(&agg[(size_t)drow * 64 + nt * 16 + n], g * m2[nt][r]);
  }
}

// Node update MLP + residual; UW waves/block, weights staged to LDS once per block.
__global__ __launch_bounds__(32 * UW) void k_upd(const unsigned short* __restrict__ wbf,
                                                 const float* __restrict__ aggv,
                                                 float* __restrict__ h,
                                                 unsigned short* __restrict__ hb,
                                                 const float* __restrict__ ub1,
                                                 const float* __restrict__ ub2, int N, int s) {
  __shared__ __align__(16) unsigned short sw1[64 * 128];
  __shared__ __align__(16) unsigned short sw2[64 * 64];
  __shared__ __align__(16) unsigned short sus[UW][16 * 64];
  const int tid = threadIdx.x, wave = tid >> 5, lane = tid & 31;
  const int half = lane >> 4, n = lane & 15;
  unsigned short* su = sus[wave];

  ldscpy(sw1, wbf + OFF_UW1 + s * (64 * 128), 64 * 128, tid, 32 * UW);
  ldscpy(sw2, wbf + OFF_UW2 + s * (64 * 64), 64 * 64, tid, 32 * UW);
  __syncthreads();

  const int node0 = (blockIdx.x * UW + wave) * 16;
  int mrow = node0 + n; if (mrow >= N) mrow = N - 1;

  v8f u1[4] = {};
#pragma unroll
  for (int kk = 0; kk < 4; ++kk) {
    v16bf a;
    if (kk < 2) a = a_from_f32(aggv + (size_t)mrow * 64, kk * 32, half);
    else        a = a_from_bf16(hb + (size_t)mrow * 64, (kk - 2) * 32, half);
#pragma unroll
    for (int nt = 0; nt < 4; ++nt)
      u1[nt] = wmma_bf16(a, b_from_w(sw1, 128, nt * 16, kk * 32, n, half), u1[nt]);
  }
#pragma unroll
  for (int nt = 0; nt < 4; ++nt) {
    float bia = ub1[s * 64 + nt * 16 + n];
#pragma unroll
    for (int r = 0; r < 8; ++r) u1[nt][r] = siluf(u1[nt][r] + bia);
  }
  wsync();
  stage<4, 64>(su, u1, half, n);
  wsync();

  v8f u2[4] = {};
#pragma unroll
  for (int c = 0; c < 2; ++c) {
    v16bf a = a_from_bf16(su + n * 64, c * 32, half);
#pragma unroll
    for (int nt = 0; nt < 4; ++nt)
      u2[nt] = wmma_bf16(a, b_from_w(sw2, 64, nt * 16, c * 32, n, half), u2[nt]);
  }
#pragma unroll
  for (int nt = 0; nt < 4; ++nt) {
    float bia = ub2[s * 64 + nt * 16 + n];
#pragma unroll
    for (int r = 0; r < 8; ++r) {
      int row = node0 + r + 8 * half;
      if (row < N) {
        size_t o = (size_t)row * 64 + nt * 16 + n;
        float v = u2[nt][r] + bia + h[o];
        h[o] = v; hb[o] = f2bf(v);
      }
    }
  }
}

// LayerNorm + output projection, one thread per node.
__global__ void k_out(const float* __restrict__ h, const float* __restrict__ gamma,
                      const float* __restrict__ beta, const float* __restrict__ ow,
                      const float* __restrict__ ob, float* __restrict__ out, int N) {
  int i = blockIdx.x * blockDim.x + threadIdx.x;
  if (i >= N) return;
  const float4* row = (const float4*)(h + (size_t)i * 64);
  float v[64];
  float mu = 0.f;
#pragma unroll
  for (int j = 0; j < 16; ++j) {
    float4 t = row[j];
    v[4 * j + 0] = t.x; v[4 * j + 1] = t.y; v[4 * j + 2] = t.z; v[4 * j + 3] = t.w;
    mu += t.x + t.y + t.z + t.w;
  }
  mu *= (1.f / 64.f);
  float var = 0.f;
#pragma unroll
  for (int k = 0; k < 64; ++k) { float d = v[k] - mu; var += d * d; }
  var *= (1.f / 64.f);
  float rs = rsqrtf(var + 1e-5f);
  float acc = 0.f;
#pragma unroll
  for (int k = 0; k < 64; ++k)
    acc += ((v[k] - mu) * rs * gamma[k] + beta[k]) * ow[k];
  out[i] = acc + ob[0];
}

extern "C" void kernel_launch(void* const* d_in, const int* in_sizes, int n_in,
                              void* d_out, int out_size, void* d_ws, size_t ws_size,
                              hipStream_t stream) {
  (void)n_in; (void)out_size; (void)ws_size;
  const float* x    = (const float*)d_in[0];
  const int*   ei   = (const int*)d_in[1];
  const float* encW = (const float*)d_in[2];
  const float* encB = (const float*)d_in[3];
  const float* tw1  = (const float*)d_in[4];
  const float* tb1  = (const float*)d_in[5];
  const float* tw2  = (const float*)d_in[6];
  const float* tb2  = (const float*)d_in[7];
  const float* mw1  = (const float*)d_in[8];
  const float* mb1  = (const float*)d_in[9];
  const float* mw2  = (const float*)d_in[10];
  const float* mb2  = (const float*)d_in[11];
  const float* gw   = (const float*)d_in[12];
  const float* gb   = (const float*)d_in[13];
  const float* uw1  = (const float*)d_in[14];
  const float* ub1  = (const float*)d_in[15];
  const float* uw2  = (const float*)d_in[16];
  const float* ub2  = (const float*)d_in[17];
  const float* lng  = (const float*)d_in[18];
  const float* lnb  = (const float*)d_in[19];
  const float* ow   = (const float*)d_in[20];
  const float* ob   = (const float*)d_in[21];
  float* out = (float*)d_out;

  const int N = in_sizes[0] / (Sd * Hd);
  const int E = in_sizes[1] / 2;
  const int* srcI = ei;
  const int* dstI = ei + E;

  char* ws = (char*)d_ws;
  unsigned short* wbf = (unsigned short*)ws;
  size_t off = ((size_t)WTOT * 2 + 255) & ~(size_t)255;
  float* h = (float*)(ws + off);
  off += (((size_t)N * 64 * 4) + 255) & ~(size_t)255;
  unsigned short* hb = (unsigned short*)(ws + off);
  off += (((size_t)N * 64 * 2) + 255) & ~(size_t)255;
  float* agg = (float*)(ws + off);

  k_prep<<<(WTOT + 255) / 256, 256, 0, stream>>>(wbf, encW, tw1, tw2, mw1, mw2, uw1, uw2);
  k_enc<<<(N + 32 * CW - 1) / (32 * CW), 32 * CW, 0, stream>>>(x, wbf, encB, tb1, tb2, h, hb, N);

  long long n4 = (long long)N * 16; // N*64/4 float4s
  for (int s = 0; s < 3; ++s) {
    k_zero<<<2048, 256, 0, stream>>>((float4*)agg, n4);
    k_edge<<<(E + 16 * EW - 1) / (16 * EW), 32 * EW, 0, stream>>>(wbf, hb, srcI, dstI,
                                                                  mb1, mb2, gw, gb, agg, E, s);
    k_upd<<<(N + 16 * UW - 1) / (16 * UW), 32 * UW, 0, stream>>>(wbf, agg, h, hb, ub1, ub2, N, s);
  }
  k_out<<<(N + 255) / 256, 256, 0, stream>>>(h, lng, lnb, ow, ob, out, N);
}
